// SirenLayerParallelMatrix_61572651155930
// MI455X (gfx1250) — matrix operand order repaired and checked by reading, not compiled, not run on hardware
//
#include <hip/hip_runtime.h>

typedef __attribute__((ext_vector_type(16))) _Float16 v16h;
typedef __attribute__((ext_vector_type(8)))  _Float16 v8h;
typedef __attribute__((ext_vector_type(4)))  _Float16 v4h;
typedef __attribute__((ext_vector_type(8)))  float    v8f;
typedef __attribute__((ext_vector_type(4)))  float    v4f;

#define DIM        128                 // block in/out width
#define NB         64                  // number of diagonal blocks
#define FEAT       (DIM * NB)          // 8192 features
#define ROWS_TOTAL 8192                // 8 * 1024 rows
#define TILE_M     256                 // rows per workgroup (8 waves x 32 rows)
#define LDS_K      (DIM + 8)           // padded f16 row stride (272 B -> conflict-free)

__launch_bounds__(256)
__global__ void siren_block_diag_wmma(const float* __restrict__ x,
                                      const float* __restrict__ w,
                                      const float* __restrict__ bias,
                                      float* __restrict__ out)
{
    __shared__ __align__(16) _Float16 xs[TILE_M * LDS_K];  // X tile, f16 [m][k]  (68 KB)
    __shared__ __align__(16) _Float16 ws[DIM    * LDS_K];  // W block, f16 [n][k] (34 KB)
    __shared__ float bs[DIM];

    const int tid = threadIdx.x;       // 0..255, 8 waves
    const int mt  = blockIdx.x;        // row tile   0..31
    const int blk = blockIdx.y;        // diag block 0..63

    const int m0 = mt  * TILE_M;
    const int f0 = blk * DIM;          // feature offset of this block

    // ---------------- stage X tile (256x128) and W block (128x128), f32 -> f16 ----------------
    {
        const float* xbase = x + (size_t)m0 * FEAT + f0;
        const float* wbase = w + (size_t)f0 * FEAT + f0;
#pragma unroll
        for (int i = 0; i < 32; ++i) {
            const int off = (tid + i * 256) * 4;   // flat f32 index in 256x128 tile
            const int r   = off >> 7;
            const int c   = off & 127;
            v4f vx = *(const v4f*)(xbase + (size_t)r * FEAT + c);
            v4h hx;
            hx.x = (_Float16)vx.x; hx.y = (_Float16)vx.y;
            hx.z = (_Float16)vx.z; hx.w = (_Float16)vx.w;
            *(v4h*)(&xs[r * LDS_K + c]) = hx;
        }
#pragma unroll
        for (int i = 0; i < 16; ++i) {
            const int off = (tid + i * 256) * 4;   // flat f32 index in 128x128 block
            const int r   = off >> 7;
            const int c   = off & 127;
            v4f vw = *(const v4f*)(wbase + (size_t)r * FEAT + c);
            v4h hw;
            hw.x = (_Float16)vw.x; hw.y = (_Float16)vw.y;
            hw.z = (_Float16)vw.z; hw.w = (_Float16)vw.w;
            *(v4h*)(&ws[r * LDS_K + c]) = hw;
        }
        if (tid < DIM) bs[tid] = bias[f0 + tid];
    }
    __syncthreads();

    // ---------------- per-wave WMMA compute: 32 rows x 128 cols per wave ----------------
    const int lane  = tid & 31;
    const int wave  = tid >> 5;        // 0..7 -> 32-row strip
    const int lmod  = lane & 15;
    const int lhalf = lane >> 4;       // 0/1
    const int mrow  = wave * 32;

    v8f acc0[8], acc1[8];
#pragma unroll
    for (int i = 0; i < 8; ++i) { v8f z = {}; acc0[i] = z; acc1[i] = z; }

#pragma unroll
    for (int ks = 0; ks < 4; ++ks) {
        const int kb = ks * 32;

        // A fragments (16x32 f16): lane half 0 holds K = kb+{0..7, 16..23},
        // half 1 holds K = kb+{8..15, 24..31}; vector elem i -> VGPR i/2, half i%2.
        const _Float16* arow0 = &xs[(mrow + lmod) * LDS_K + kb + lhalf * 8];
        const _Float16* arow1 = arow0 + 16 * LDS_K;
        v8h a0lo = *(const v8h*)(arow0);
        v8h a0hi = *(const v8h*)(arow0 + 16);
        v8h a1lo = *(const v8h*)(arow1);
        v8h a1hi = *(const v8h*)(arow1 + 16);
        v16h afr0, afr1;
#pragma unroll
        for (int i = 0; i < 8; ++i) {
            afr0[i] = a0lo[i]; afr0[i + 8] = a0hi[i];
            afr1[i] = a1lo[i]; afr1[i + 8] = a1hi[i];
        }

#pragma unroll
        for (int nt = 0; nt < 8; ++nt) {
            // B fragment (32x16 f16): lane holds column n = nt*16+lmod,
            // K = kb + lhalf*8 + {0..7, 16..23}.  B[k,n] = W[n,k] -> row
            // segments of W: two ds_load_b128, no transpose needed.
            const _Float16* brow = &ws[(nt * 16 + lmod) * LDS_K + kb + lhalf * 8];
            v8h b_lo = *(const v8h*)(brow);
            v8h b_hi = *(const v8h*)(brow + 16);
            v16h bfr;
#pragma unroll
            for (int i = 0; i < 8; ++i) { bfr[i] = b_lo[i]; bfr[i + 8] = b_hi[i]; }

            acc0[nt] = __builtin_amdgcn_wmma_f32_16x16x32_f16(
                false, afr0, false, bfr, (short)0, acc0[nt], false, false);
            acc1[nt] = __builtin_amdgcn_wmma_f32_16x16x32_f16(
                false, afr1, false, bfr, (short)0, acc1[nt], false, false);
        }
    }

    // ---------------- epilogue: bias + wire activation + store ----------------
    // C/D layout: VGPR r, lanes 0-15 -> (M=r, N=lane), lanes 16-31 -> (M=8+r, N=lane-16)
#pragma unroll
    for (int s = 0; s < 2; ++s) {
        float* obase = out + (size_t)(m0 + mrow + s * 16 + lhalf * 8) * FEAT + f0;
#pragma unroll
        for (int nt = 0; nt < 8; ++nt) {
            const int   n  = nt * 16 + lmod;
            const float bv = bs[n];
#pragma unroll
            for (int r = 0; r < 8; ++r) {
                float zv = (s ? acc1[nt][r] : acc0[nt][r]) + bv;
                float sz = 0.1f * zv;
                obase[(size_t)r * FEAT + n] = __sinf(30.0f * zv) * __expf(-sz * sz);
            }
        }
    }
}

extern "C" void kernel_launch(void* const* d_in, const int* in_sizes, int n_in,
                              void* d_out, int out_size, void* d_ws, size_t ws_size,
                              hipStream_t stream) {
    const float* x    = (const float*)d_in[0];
    const float* wgt  = (const float*)d_in[1];
    const float* bias = (const float*)d_in[2];
    float* out        = (float*)d_out;

    dim3 grid(ROWS_TOTAL / TILE_M, NB);   // 32 x 64 workgroups
    siren_block_diag_wmma<<<grid, 256, 0, stream>>>(x, wgt, bias, out);
}
